// GatedMultiHeadAttention_8022998909000
// MI455X (gfx1250) — matrix operand order repaired and checked by reading, not compiled, not run on hardware
//
#include <hip/hip_runtime.h>
#include <hip/hip_bf16.h>

#define D_MODELC 1024
#define NUM_HEADSC 16
#define D_KC 64
#define BATCHC 2
#define SEQC 2048
#define MROWSC (BATCHC * SEQC) /* 4096 */

typedef __bf16 bf16;
typedef __attribute__((ext_vector_type(8)))  __bf16 v8bf;
typedef __attribute__((ext_vector_type(16))) __bf16 v16bf;
typedef __attribute__((ext_vector_type(8)))  float  v8f;
typedef __attribute__((ext_vector_type(4)))  int    v4i;

// ---------------------------------------------------------------------------
// Async global->LDS staging (CDNA5 GLOBAL_LOAD_ASYNC_TO_LDS_B128, ASYNCcnt).
// Signature (probe-verified): (int4 AS1* src, int4 AS3* dst, imm off, imm cpol)
// ---------------------------------------------------------------------------
#if __has_builtin(__builtin_amdgcn_global_load_async_to_lds_b128) && \
    __has_builtin(__builtin_amdgcn_s_wait_asynccnt)
#define HAVE_ASYNC_LDS 1
#else
#define HAVE_ASYNC_LDS 0
#endif

typedef __attribute__((address_space(1))) v4i gbl_v4i;
typedef __attribute__((address_space(3))) v4i lds_v4i;

// copy 16 bytes (8 bf16) from global to LDS, per lane
static __device__ inline void copy_b128(bf16* lds_dst, const bf16* gsrc) {
#if HAVE_ASYNC_LDS
  __builtin_amdgcn_global_load_async_to_lds_b128(
      (gbl_v4i*)(unsigned long long)gsrc,
      (lds_v4i*)(unsigned int)(unsigned long long)lds_dst, 0, 0);
#else
  *(v8bf*)lds_dst = *(const v8bf*)gsrc;
#endif
}

static __device__ inline void stage_wait() {
#if HAVE_ASYNC_LDS
  __builtin_amdgcn_s_wait_asynccnt(0);
#endif
}

static __device__ inline v8f zero8() {
  v8f z;
#pragma unroll
  for (int i = 0; i < 8; ++i) z[i] = 0.0f;
  return z;
}

static __device__ inline v16bf pack_a(v8bf lo, v8bf hi) {
  v16bf a;
#pragma unroll
  for (int i = 0; i < 8; ++i) { a[i] = lo[i]; a[i + 8] = hi[i]; }
  return a;
}

// A-fragment (16x32 bf16, MxK): lane elements k = half*8 + j (j<8),
// k = 16 + half*8 + (j-8) (j>=8); base already points at row m = lane&15.
static __device__ inline v16bf load_a16(const bf16* base, int half) {
  v8bf lo = *(const v8bf*)(base + half * 8);
  v8bf hi = *(const v8bf*)(base + 16 + half * 8);
  return pack_a(lo, hi);
}

// B-fragment helper when the 16 k-values for this lane are contiguous.
static __device__ inline v16bf load_row16(const bf16* base) {
  v8bf lo = *(const v8bf*)(base);
  v8bf hi = *(const v8bf*)(base + 8);
  return pack_a(lo, hi);
}

static __device__ inline v16bf ones16() {
  v16bf o;
#pragma unroll
  for (int i = 0; i < 16; ++i) o[i] = (bf16)1.0f;
  return o;
}

static __device__ inline v8f wmma_bf16(v16bf a, v16bf b, v8f c) {
  return __builtin_amdgcn_wmma_f32_16x16x32_bf16(false, a, false, b, (short)0, c,
                                                 false, false);
}

// ---------------------------------------------------------------------------
// DPP16 butterfly max-reduction across each 16-lane group (VALU only).
// ---------------------------------------------------------------------------
static __device__ inline float dpp_max16(float v) {
  int t;
  t = __builtin_amdgcn_update_dpp(__float_as_int(v), __float_as_int(v), 0xB1, 0xF, 0xF, false);
  v = fmaxf(v, __int_as_float(t));
  t = __builtin_amdgcn_update_dpp(__float_as_int(v), __float_as_int(v), 0x4E, 0xF, 0xF, false);
  v = fmaxf(v, __int_as_float(t));
  t = __builtin_amdgcn_update_dpp(__float_as_int(v), __float_as_int(v), 0x141, 0xF, 0xF, false);
  v = fmaxf(v, __int_as_float(t));
  t = __builtin_amdgcn_update_dpp(__float_as_int(v), __float_as_int(v), 0x140, 0xF, 0xF, false);
  v = fmaxf(v, __int_as_float(t));
  return v;
}

// ---------------------------------------------------------------------------
// fp32 -> bf16 elementwise conversion
// ---------------------------------------------------------------------------
__global__ void cvt_kernel(const float* __restrict__ in, bf16* __restrict__ out,
                           int n) {
  int i = blockIdx.x * blockDim.x + threadIdx.x;
  if (i < n) out[i] = (bf16)in[i];
}

// ---------------------------------------------------------------------------
// fp32 [K][N] -> bf16 transposed [N][K], tiled 32x32 through LDS (coalesced).
// ---------------------------------------------------------------------------
__global__ __launch_bounds__(256) void cvt_t_kernel(const float* __restrict__ in,
                                                    bf16* __restrict__ out) {
  __shared__ float tile[32][33];
  const int tx = threadIdx.x, ty = threadIdx.y;
  const int k0 = blockIdx.y * 32, n0 = blockIdx.x * 32;
#pragma unroll
  for (int i = 0; i < 4; ++i)
    tile[ty + i * 8][tx] = in[(size_t)(k0 + ty + i * 8) * D_MODELC + n0 + tx];
  __syncthreads();
#pragma unroll
  for (int i = 0; i < 4; ++i)
    out[(size_t)(n0 + ty + i * 8) * D_MODELC + k0 + tx] =
        (bf16)tile[tx][ty + i * 8];
}

// ---------------------------------------------------------------------------
// Tiled bf16 WMMA GEMM: C[M,N] = A[M,K] * Bt[N,K]^T + bias, 64-deep k-chunks,
// double-buffered async LDS staging. Block = 128 threads (4 waves), 64x64
// tile; each wave a 32x32 sub-tile; 8 WMMAs per barrier round.
// MODE: 0 = f32 row-major out, 1 = bf16 row-major out,
//       2 = bf16 out written V-transposed as [b*1024+col][SEQ] (for V).
// ---------------------------------------------------------------------------
template <int MODE>
__global__ __launch_bounds__(128) void gemm_kernel(
    const bf16* __restrict__ A, const bf16* __restrict__ Bt,
    const float* __restrict__ bias, void* __restrict__ Cout, int M, int Kd,
    int Nd) {
  __shared__ __align__(16) bf16 Xl[2][64 * 64];
  __shared__ __align__(16) bf16 Wtl[2][64 * 64];

  const int lane = threadIdx.x & 31;
  const int wave = threadIdx.x >> 5;
  const int half = lane >> 4;
  const int l15 = lane & 15;
  const int mblk = blockIdx.y * 64;
  const int nblk = blockIdx.x * 64;
  const int mw = (wave >> 1) * 32;
  const int nw = (wave & 1) * 32;

  const int t = threadIdx.x;
  const int xr = t >> 1;          // 0..63
  const int xo = (t & 1) * 32;    // 0,32
  const bf16* agp = A + (size_t)(mblk + xr) * Kd + xo;
  const bf16* bgp = Bt + (size_t)(nblk + xr) * Kd + xo;

  auto stage = [&](int kk, int buf) {
#pragma unroll
    for (int c = 0; c < 4; ++c) {
      copy_b128(&Xl[buf][xr * 64 + xo + c * 8], agp + kk + c * 8);
      copy_b128(&Wtl[buf][xr * 64 + xo + c * 8], bgp + kk + c * 8);
    }
#if !HAVE_ASYNC_LDS
    if (kk + 64 < Kd) {
      __builtin_prefetch((const void*)(agp + kk + 64), 0, 3);
      __builtin_prefetch((const void*)(bgp + kk + 64), 0, 3);
    }
#endif
  };

  v8f acc[2][2];
#pragma unroll
  for (int mi = 0; mi < 2; ++mi)
#pragma unroll
    for (int ni = 0; ni < 2; ++ni) acc[mi][ni] = zero8();

  stage(0, 0);
  for (int kk = 0, it = 0; kk < Kd; kk += 64, ++it) {
    const int cur = it & 1;
    stage_wait();
    __syncthreads();
    if (kk + 64 < Kd) stage(kk + 64, cur ^ 1);

    v16bf a[2][2];  // [mi][k2]
#pragma unroll
    for (int mi = 0; mi < 2; ++mi)
#pragma unroll
      for (int k2 = 0; k2 < 2; ++k2)
        a[mi][k2] =
            load_a16(&Xl[cur][(mw + mi * 16 + l15) * 64 + k2 * 32], half);
#pragma unroll
    for (int ni = 0; ni < 2; ++ni) {
#pragma unroll
      for (int k2 = 0; k2 < 2; ++k2) {
        const v16bf bfr = load_a16(
            &Wtl[cur][(nw + ni * 16 + l15) * 64 + k2 * 32], half);
        acc[0][ni] = wmma_bf16(a[0][k2], bfr, acc[0][ni]);
        acc[1][ni] = wmma_bf16(a[1][k2], bfr, acc[1][ni]);
      }
    }
  }

#pragma unroll
  for (int ni = 0; ni < 2; ++ni) {
    const int col = nblk + nw + ni * 16 + l15;
    const float bval = bias[col];
#pragma unroll
    for (int mi = 0; mi < 2; ++mi) {
      if (MODE == 2) {
        // V^T layout: row index (b*1024 + col), token consecutive over r.
        const int row0 = mblk + mw + mi * 16 + half * 8;
        const int bb = row0 >> 11;
        const int tok = row0 & (SEQC - 1);
        v8bf pk;
#pragma unroll
        for (int r = 0; r < 8; ++r) pk[r] = (bf16)(acc[mi][ni][r] + bval);
        *(v8bf*)&((bf16*)Cout)[((size_t)bb * 1024 + col) * SEQC + tok] = pk;
      } else {
#pragma unroll
        for (int r = 0; r < 8; ++r) {
          const int row = mblk + mw + mi * 16 + r + half * 8;
          const float v = acc[mi][ni][r] + bval;
          if (MODE == 1)
            ((bf16*)Cout)[(size_t)row * Nd + col] = (bf16)v;
          else
            ((float*)Cout)[(size_t)row * Nd + col] = v;
        }
      }
    }
  }
}

// ---------------------------------------------------------------------------
// Head-wise sigmoid gate: g[row,h] = sigmoid(x[row,:] . Wg[:,h] + bg[h])
// ---------------------------------------------------------------------------
__global__ void gate_kernel(const float* __restrict__ x,
                            const float* __restrict__ Wg,
                            const float* __restrict__ bg,
                            float* __restrict__ g) {
  const int t = blockIdx.x * blockDim.x + threadIdx.x;  // 65536 total
  const int row = t >> 4;
  const int h = t & 15;
  float acc = bg[h];
  const float* xr = x + (size_t)row * D_MODELC;
  for (int k = 0; k < D_MODELC; ++k) acc += xr[k] * Wg[k * NUM_HEADSC + h];
  g[t] = 1.0f / (1.0f + __expf(-acc));
}

// ---------------------------------------------------------------------------
// Flash attention, 64-key blocks, double-buffered async K/V^T staging.
// V is TRANSPOSED: Vt[(b*1024 + h*64 + d)][SEQ]. Row sums of P are computed
// with WMMA against an all-ones B matrix (P*1 broadcasts row sums to all
// lanes), replacing 8 serialized DPP sum chains with 2 matrix ops.
// ---------------------------------------------------------------------------
__global__ __launch_bounds__(128) void attn_kernel(
    const bf16* __restrict__ Q, const bf16* __restrict__ K,
    const bf16* __restrict__ Vt, const float* __restrict__ G,
    bf16* __restrict__ Y) {
  __shared__ __align__(16) bf16 Klds[2][64 * 64];
  __shared__ __align__(16) bf16 Vtl[2][64 * 64];
  __shared__ __align__(16) bf16 Plds[4][16 * 64];

  const int lane = threadIdx.x & 31;
  const int wave = threadIdx.x >> 5;
  const int half = lane >> 4;
  const int l15 = lane & 15;
  const int b = blockIdx.y >> 4;
  const int h = blockIdx.y & 15;
  const int qbase = blockIdx.x * 64 + wave * 16;
  const size_t rowbase = (size_t)b * SEQC;

  const int t = threadIdx.x;
  const int srow = t >> 1;        // 0..63 (key row for K; d row for Vt)
  const int soff = (t & 1) * 32;  // 0,32
  const bf16* kgp = K + (rowbase + srow) * D_MODELC + h * D_KC + soff;
  const bf16* vgp = Vt + ((size_t)b * 1024 + h * 64 + srow) * SEQC + soff;

  auto stage = [&](int kb, int buf) {
    const bf16* ks = kgp + (size_t)kb * 64 * D_MODELC;
    const bf16* vs = vgp + kb * 64;
#pragma unroll
    for (int c = 0; c < 4; ++c) {
      copy_b128(&Klds[buf][srow * 64 + soff + c * 8], ks + c * 8);
      copy_b128(&Vtl[buf][srow * 64 + soff + c * 8], vs + c * 8);
    }
#if !HAVE_ASYNC_LDS
    if (kb + 1 < SEQC / 64) {
      __builtin_prefetch((const void*)(ks + (size_t)64 * D_MODELC), 0, 3);
      __builtin_prefetch((const void*)(vs + 64), 0, 3);
    }
#endif
  };

  // Q tile A-fragments (reused for all key blocks), k = 0..31 and 32..63.
  const bf16* qrow = Q + (rowbase + qbase + l15) * D_MODELC + h * D_KC;
  const v16bf aq0 = load_a16(qrow, half);
  const v16bf aq1 = load_a16(qrow + 32, half);
  const v16bf one_b = ones16();

  float m_run[8], l_run[8];
  v8f O[4];
#pragma unroll
  for (int r = 0; r < 8; ++r) { m_run[r] = -3.0e38f; l_run[r] = 0.0f; }
#pragma unroll
  for (int nt = 0; nt < 4; ++nt) O[nt] = zero8();

  stage(0, 0);
  for (int kb = 0; kb < SEQC / 64; ++kb) {
    const int cur = kb & 1;
    stage_wait();
    __syncthreads();
    if (kb + 1 < SEQC / 64) stage(kb + 1, cur ^ 1);

    // scores: S[s] covers keys s*16 .. s*16+15 of this 64-key block
    v8f S[4];
    {
      const bf16* kl = &Klds[cur][0];
#pragma unroll
      for (int s = 0; s < 4; ++s) {
        v8f c = zero8();
        c = wmma_bf16(aq0, load_a16(kl + (s * 16 + l15) * 64, half), c);
        c = wmma_bf16(aq1,
                      load_a16(kl + (s * 16 + l15) * 64 + 32, half), c);
        S[s] = c;
      }
    }

    bf16* pw = &Plds[wave][0];
    float alpha_a[8];
#pragma unroll
    for (int r = 0; r < 8; ++r) {
      float sv[4];
#pragma unroll
      for (int s = 0; s < 4; ++s) sv[s] = S[s][r] * 0.125f;  // 1/sqrt(64)
      const float mb =
          dpp_max16(fmaxf(fmaxf(sv[0], sv[1]), fmaxf(sv[2], sv[3])));
      const float mn = fmaxf(m_run[r], mb);
      alpha_a[r] = __expf(m_run[r] - mn);
      m_run[r] = mn;
#pragma unroll
      for (int nt = 0; nt < 4; ++nt) O[nt][r] *= alpha_a[r];
#pragma unroll
      for (int s = 0; s < 4; ++s) {
        const float pv = __expf(sv[s] - mn);
        pw[(r + half * 8) * 64 + s * 16 + l15] = (bf16)pv;
      }
    }

    // P as two A-fragments (wave-private LDS; DS in-order within a wave)
    const v16bf aP0 = load_a16(&pw[l15 * 64], half);
    const v16bf aP1 = load_a16(&pw[l15 * 64 + 32], half);

    // block row-sums via WMMA: (P * ones) broadcasts row sums to all lanes
    v8f lsum = zero8();
    lsum = wmma_bf16(aP0, one_b, lsum);
    lsum = wmma_bf16(aP1, one_b, lsum);
#pragma unroll
    for (int r = 0; r < 8; ++r) l_run[r] = l_run[r] * alpha_a[r] + lsum[r];

#pragma unroll
    for (int nt = 0; nt < 4; ++nt) {
      const bf16* vl = &Vtl[cur][(nt * 16 + l15) * 64];
      O[nt] = wmma_bf16(aP0, load_a16(vl, half), O[nt]);
      O[nt] = wmma_bf16(aP1, load_a16(vl + 32, half), O[nt]);
    }
  }

  // epilogue: 1/l normalization * sigmoid gate, write Y (merged-head layout)
#pragma unroll
  for (int r = 0; r < 8; ++r) {
    const int row = qbase + r + half * 8;
    const float g = G[(rowbase + row) * NUM_HEADSC + h];
    const float sc = g / l_run[r];
#pragma unroll
    for (int nt = 0; nt < 4; ++nt) {
      Y[(rowbase + row) * (size_t)D_MODELC + h * D_KC + nt * 16 + l15] =
          (bf16)(O[nt][r] * sc);
    }
  }
}

// ---------------------------------------------------------------------------
extern "C" void kernel_launch(void* const* d_in, const int* in_sizes, int n_in,
                              void* d_out, int out_size, void* d_ws,
                              size_t ws_size, hipStream_t stream) {
  (void)in_sizes; (void)n_in; (void)out_size; (void)ws_size;
  const float* x  = (const float*)d_in[0];
  const float* Wq = (const float*)d_in[1];
  const float* bq = (const float*)d_in[2];
  const float* Wk = (const float*)d_in[3];
  const float* bk = (const float*)d_in[4];
  const float* Wv = (const float*)d_in[5];
  const float* bv = (const float*)d_in[6];
  const float* Wg = (const float*)d_in[7];
  const float* bg = (const float*)d_in[8];
  const float* Wo = (const float*)d_in[9];
  const float* bo = (const float*)d_in[10];
  float* out = (float*)d_out;

  char* p = (char*)d_ws;
  auto alloc = [&](size_t bytes) {
    char* r = p;
    p += (bytes + 255) & ~(size_t)255;
    return r;
  };
  const size_t act_elems = (size_t)MROWSC * D_MODELC;  // 4096*1024
  const size_t w_elems = (size_t)D_MODELC * D_MODELC;  // 1024*1024
  bf16* xb   = (bf16*)alloc(act_elems * sizeof(bf16));
  bf16* wqt  = (bf16*)alloc(w_elems * sizeof(bf16));
  bf16* wkt  = (bf16*)alloc(w_elems * sizeof(bf16));
  bf16* wvt  = (bf16*)alloc(w_elems * sizeof(bf16));
  bf16* wot  = (bf16*)alloc(w_elems * sizeof(bf16));
  bf16* qb   = (bf16*)alloc(act_elems * sizeof(bf16));
  bf16* kbuf = (bf16*)alloc(act_elems * sizeof(bf16));
  bf16* vtb  = (bf16*)alloc(act_elems * sizeof(bf16));
  bf16* yb   = (bf16*)alloc(act_elems * sizeof(bf16));
  float* gbuf = (float*)alloc((size_t)MROWSC * NUM_HEADSC * sizeof(float));

  cvt_kernel<<<(int)(act_elems / 256), 256, 0, stream>>>(x, xb, (int)act_elems);
  dim3 tgrid(D_MODELC / 32, D_MODELC / 32);  // (32,32)
  dim3 tblk(32, 8);
  cvt_t_kernel<<<tgrid, tblk, 0, stream>>>(Wq, wqt);
  cvt_t_kernel<<<tgrid, tblk, 0, stream>>>(Wk, wkt);
  cvt_t_kernel<<<tgrid, tblk, 0, stream>>>(Wv, wvt);
  cvt_t_kernel<<<tgrid, tblk, 0, stream>>>(Wo, wot);

  dim3 ggrid(D_MODELC / 64, MROWSC / 64);  // (16, 64)
  gemm_kernel<1><<<ggrid, 128, 0, stream>>>(xb, wqt, bq, qb, MROWSC, D_MODELC,
                                            D_MODELC);
  gemm_kernel<1><<<ggrid, 128, 0, stream>>>(xb, wkt, bk, kbuf, MROWSC, D_MODELC,
                                            D_MODELC);
  gemm_kernel<2><<<ggrid, 128, 0, stream>>>(xb, wvt, bv, vtb, MROWSC, D_MODELC,
                                            D_MODELC);

  gate_kernel<<<(MROWSC * NUM_HEADSC) / 256, 256, 0, stream>>>(x, Wg, bg, gbuf);

  attn_kernel<<<dim3(SEQC / 64, BATCHC * NUM_HEADSC), 128, 0, stream>>>(
      qb, kbuf, vtb, gbuf, yb);

  gemm_kernel<0><<<ggrid, 128, 0, stream>>>(yb, wot, bo, out, MROWSC, D_MODELC,
                                            D_MODELC);
}
